// MultiHeadSelfAttention_57062935495565
// MI455X (gfx1250) — matrix operand order repaired and checked by reading, not compiled, not run on hardware
//
#include <hip/hip_runtime.h>

// MI455X / gfx1250 multi-head self-attention.
// f16 WMMA (v_wmma_f32_16x16x32_f16) everywhere, f32 accumulate.
// Compute-bound (~103 GFLOP vs ~2.2us HBM traffic at 23.3 TB/s).
// GEMMs: TDM (tensor_load_to_lds) double-buffered LDS staging, TENSORcnt
// synchronized; fragments come from conflict-free padded LDS (stride 72 halves
// produced by the TDM's hardware padding).
// Attention: flash-style, barrier-free main loop, V^T produced at GEMM time,
// softmax butterflies on v_permlane16 (VALU co-executes with XDL WMMA).

typedef __attribute__((ext_vector_type(16))) _Float16 v16h;
typedef __attribute__((ext_vector_type(8)))  _Float16 v8h;
typedef __attribute__((ext_vector_type(8)))  float    v8f;
typedef __attribute__((ext_vector_type(4)))  float    v4f;
typedef __attribute__((ext_vector_type(4)))  _Float16 v4h;
typedef unsigned int u32x4 __attribute__((ext_vector_type(4)));
typedef unsigned int u32x8 __attribute__((ext_vector_type(8)));

#define D_MODEL 1024
#define N_HEADS 16
#define HEAD_DIM 64
#define BATCH 2
#define SEQ 2048
#define NTOK (BATCH * SEQ) /* 4096 */

#if __has_builtin(__builtin_amdgcn_exp2f)
#define EXP2(x) __builtin_amdgcn_exp2f(x)
#else
#define EXP2(x) exp2f(x)
#endif

static __device__ inline v8f wmma32(v16h a, v16h b, v8f c) {
  return __builtin_amdgcn_wmma_f32_16x16x32_f16(
      /*neg_a=*/false, a, /*neg_b=*/false, b,
      /*c_mod=*/(short)0, c, /*reuse_a=*/false, /*reuse_b=*/false);
}

// A fragment: 16x32 f16, p0 -> (row0, k0) of row-major [*, ld].
// Lane<16: row=lane, halves 0..7 = K+0..7, halves 8..15 = K+16..23.
// Lane>=16: same rows, K ranges shifted by +8 (ISA 7.12.2).
static __device__ inline v16h frag_a(const _Float16* p0, int ld) {
  const int lane = threadIdx.x & 31;
  const _Float16* p = p0 + (size_t)(lane & 15) * ld + ((lane >> 4) << 3);
  v8h lo = *(const v8h*)p;
  v8h hi = *(const v8h*)(p + 16);
  v16h r;
#pragma unroll
  for (int i = 0; i < 8; ++i) { r[i] = lo[i]; r[i + 8] = hi[i]; }
  return r;
}

// B fragment: 32x16 f16 with B[k][n] = SRC[n][k] (SRC row-major [*, ld]),
// p0 -> (n0, k0). Lane<16: col n=lane, halves 0..7 = K+0..7, 8..15 = K+16..23; lanes>=16: +8.
static __device__ inline v16h frag_b(const _Float16* p0, int ld) {
  const int lane = threadIdx.x & 31;
  const _Float16* p = p0 + (size_t)(lane & 15) * ld + ((lane >> 4) << 3);
  v8h lo = *(const v8h*)p;
  v8h hi = *(const v8h*)(p + 16);
  v16h r;
#pragma unroll
  for (int i = 0; i < 8; ++i) { r[i] = lo[i]; r[i + 8] = hi[i]; }
  return r;
}

// ---------------------------------------------------------------------------
// Intra-16-lane butterfly reductions on v_permlane16_b32 (pure VALU,
// co-executes with XDL-tracked WMMA, keeps LDS free for fragment traffic).
#if __has_builtin(__builtin_amdgcn_permlane16)
static __device__ inline float plxor(float x, unsigned lo, unsigned hi) {
  int v = __builtin_bit_cast(int, x);
  v = __builtin_amdgcn_permlane16(v, v, lo, hi, false, false);
  return __builtin_bit_cast(float, v);
}
static __device__ inline float redmax16(float x) {
  x = fmaxf(x, plxor(x, 0x67452301u, 0xEFCDAB89u));  // ^1
  x = fmaxf(x, plxor(x, 0x54761032u, 0xDCFE98BAu));  // ^2
  x = fmaxf(x, plxor(x, 0x32107654u, 0xBA98FEDCu));  // ^4
  x = fmaxf(x, plxor(x, 0xFEDCBA98u, 0x76543210u));  // ^8
  return x;
}
static __device__ inline float redsum16(float x) {
  x += plxor(x, 0x67452301u, 0xEFCDAB89u);
  x += plxor(x, 0x54761032u, 0xDCFE98BAu);
  x += plxor(x, 0x32107654u, 0xBA98FEDCu);
  x += plxor(x, 0xFEDCBA98u, 0x76543210u);
  return x;
}
#else
static __device__ inline float redmax16(float x) {
  x = fmaxf(x, __shfl_xor(x, 1, 32));
  x = fmaxf(x, __shfl_xor(x, 2, 32));
  x = fmaxf(x, __shfl_xor(x, 4, 32));
  x = fmaxf(x, __shfl_xor(x, 8, 32));
  return x;
}
static __device__ inline float redsum16(float x) {
  x += __shfl_xor(x, 1, 32);
  x += __shfl_xor(x, 2, 32);
  x += __shfl_xor(x, 4, 32);
  x += __shfl_xor(x, 8, 32);
  return x;
}
#endif

// ---------------------------------------------------------------------------
// TDM: async tensor DMA global -> LDS (ISA ch. 7/8 of CDNA5 ISA).
// 2-D tile of f16: tile_dim0 = 64 halves (row segment), tile_dim1 = rows.
// Hardware padding: pad_interval = 32 dwords (64 halves), pad_amount = 4
// dwords (8 halves) -> LDS row stride 72 halves (144 B): 16-B aligned and
// bank-conflict-free for b128 fragment reads (36-dword lane stride).
typedef __attribute__((address_space(3))) _Float16 lds_h;
static __device__ inline unsigned lds_off(_Float16* p) {
  return (unsigned)(uintptr_t)(lds_h*)p;
}

static __device__ inline void tdm_load_tile(unsigned lds_byte_off,
                                            const _Float16* gsrc,
                                            int tile_rows, int tensor_rows) {
  const unsigned long long ga = (unsigned long long)(uintptr_t)gsrc;
  u32x4 g0;
  g0.x = 1u;                                   // count=1 (valid descriptor)
  g0.y = lds_byte_off;                         // lds_addr
  g0.z = (unsigned)ga;                         // global_addr[31:0]
  g0.w = ((unsigned)(ga >> 32) & 0x01FFFFFFu)  // global_addr[56:32]
         | 0x80000000u;                        // type=2 ("image")
  u32x8 g1;
  g1[0] = 0x10000u        // data_size=1 (2 bytes)
          | (1u << 20)    // pad_enable
          | (4u << 22)    // pad_interval: 32 dwords
          | (3u << 25);   // pad_amount: 4 dwords
  g1[1] = (unsigned)D_MODEL << 16;      // tensor_dim0 (low16) in [31:16]
  g1[2] = (unsigned)tensor_rows << 16;  // tensor_dim0 hi=0 | tensor_dim1 lo16
  g1[3] = 64u << 16;                    // tensor_dim1 hi=0 | tile_dim0=64
  g1[4] = (unsigned)tile_rows;          // tile_dim1 | tile_dim2=0
  g1[5] = (unsigned)D_MODEL;            // tensor_dim0_stride[31:0]
  g1[6] = 0u;                           // stride0 hi | tensor_dim1_stride lo
  g1[7] = 0u;                           // tensor_dim1_stride hi
  asm volatile("tensor_load_to_lds %0, %1" ::"s"(g0), "s"(g1) : "memory");
}

// ---------------------------------------------------------------------------
// f32 -> f16 cast, 4 elements per thread
__global__ __launch_bounds__(256) void cast_f32_f16(const float* __restrict__ s,
                                                    _Float16* __restrict__ d,
                                                    int n) {
  int i = (blockIdx.x * 256 + threadIdx.x) * 4;
  if (i < n) {
    v4f v = *(const v4f*)(s + i);
    v4h h;
#pragma unroll
    for (int j = 0; j < 4; ++j) h[j] = (_Float16)v[j];
    *(v4h*)(d + i) = h;
  }
}

// ---------------------------------------------------------------------------
// Y = X[N,1024](f16) * W[1024,1024]^T (W stored [out,in]).
// Block: 256 thr = 8 waves, tile 128(M) x 64(N); wave tile 32x32 (4 accs).
// K loop: stages of 64, TDM double-buffered through LDS; wave 0 drives the
// DMA, TENSORcnt + block barrier hand off buffers.
// MODE: 0 = f16 row-major, 1 = f32 row-major (final output),
//       2 = f16 transposed-per-head V^T store: VT[b][h][hd][s].
enum { OUT_F16 = 0, OUT_F32 = 1, OUT_VT = 2 };
#define KST 64
#define LSTR 72 /* halves; set by TDM pad fields */

static __device__ inline size_t vt_idx(int r, int c) {
  // token r = b*SEQ + s ; channel c = h*64 + n  ->  ((b*H + h)*64 + n)*SEQ + s
  return ((size_t)((r >> 11) * N_HEADS + (c >> 6)) * HEAD_DIM + (c & 63)) * SEQ +
         (size_t)(r & (SEQ - 1));
}

template <int MODE>
__global__ __launch_bounds__(256) void gemm_xwt(const _Float16* __restrict__ X,
                                                const _Float16* __restrict__ W,
                                                void* __restrict__ Yv) {
  __shared__ __attribute__((aligned(16))) _Float16 Xl[2][128 * LSTR];
  __shared__ __attribute__((aligned(16))) _Float16 Wl[2][64 * LSTR];

  const int wid = threadIdx.x >> 5;
  const int lane = threadIdx.x & 31;
  const int rowblk = blockIdx.y * 128;
  const int colblk = blockIdx.x * 64;
  const int wm = wid >> 1, wn = wid & 1;

  if (wid == 0) {  // prologue: stage 0 into buffer 0
    tdm_load_tile(lds_off(&Xl[0][0]), X + (size_t)rowblk * D_MODEL, 128, NTOK);
    tdm_load_tile(lds_off(&Wl[0][0]), W + (size_t)colblk * D_MODEL, 64, D_MODEL);
  }

  v8f a00 = {}, a01 = {}, a10 = {}, a11 = {};
  for (int st = 0; st < D_MODEL / KST; ++st) {
    const int buf = st & 1;
    if (wid == 0) __builtin_amdgcn_s_wait_tensorcnt(0);
    __syncthreads();  // buffer `buf` filled; buffer buf^1 free of readers
    if (wid == 0 && st + 1 < D_MODEL / KST) {
      const int k0n = (st + 1) * KST;
      tdm_load_tile(lds_off(&Xl[buf ^ 1][0]),
                    X + (size_t)rowblk * D_MODEL + k0n, 128, NTOK);
      tdm_load_tile(lds_off(&Wl[buf ^ 1][0]),
                    W + (size_t)colblk * D_MODEL + k0n, 64, D_MODEL);
    }
#pragma unroll
    for (int kk = 0; kk < KST; kk += 32) {
      v16h fa0 = frag_a(&Xl[buf][(wm * 32) * LSTR + kk], LSTR);
      v16h fa1 = frag_a(&Xl[buf][(wm * 32 + 16) * LSTR + kk], LSTR);
      v16h fb0 = frag_b(&Wl[buf][(wn * 32) * LSTR + kk], LSTR);
      v16h fb1 = frag_b(&Wl[buf][(wn * 32 + 16) * LSTR + kk], LSTR);
      a00 = wmma32(fa0, fb0, a00);
      a01 = wmma32(fa0, fb1, a01);
      a10 = wmma32(fa1, fb0, a10);
      a11 = wmma32(fa1, fb1, a11);
    }
  }

  const int row0 = rowblk + wm * 32;
  const int col0 = colblk + wn * 32;
  const int col = lane & 15;
  const int rb = (lane >> 4) << 3;
#pragma unroll
  for (int i = 0; i < 8; ++i) {
    const int r0 = row0 + rb + i, r1 = row0 + 16 + rb + i;
    const int c0 = col0 + col, c1 = col0 + 16 + col;
    if constexpr (MODE == OUT_F32) {
      float* Y = (float*)Yv;
      Y[(size_t)r0 * D_MODEL + c0] = a00[i];
      Y[(size_t)r0 * D_MODEL + c1] = a01[i];
      Y[(size_t)r1 * D_MODEL + c0] = a10[i];
      Y[(size_t)r1 * D_MODEL + c1] = a11[i];
    } else if constexpr (MODE == OUT_VT) {
      _Float16* Y = (_Float16*)Yv;
      Y[vt_idx(r0, c0)] = (_Float16)a00[i];
      Y[vt_idx(r0, c1)] = (_Float16)a01[i];
      Y[vt_idx(r1, c0)] = (_Float16)a10[i];
      Y[vt_idx(r1, c1)] = (_Float16)a11[i];
    } else {
      _Float16* Y = (_Float16*)Yv;
      Y[(size_t)r0 * D_MODEL + c0] = (_Float16)a00[i];
      Y[(size_t)r0 * D_MODEL + c1] = (_Float16)a01[i];
      Y[(size_t)r1 * D_MODEL + c0] = (_Float16)a10[i];
      Y[(size_t)r1 * D_MODEL + c1] = (_Float16)a11[i];
    }
  }
}

// ---------------------------------------------------------------------------
// Flash attention. Grid: (B*H, S/64). Block: 128 thr = 4 independent waves,
// each owning 16 query rows. KV streamed in causal chunks of 64 keys.
// K fragments and V^T fragments come straight from global (L2-resident);
// the only LDS use is the per-wave 16x64 P tile -> no block barriers.
__global__ __launch_bounds__(128) void flash_attn(const _Float16* __restrict__ Q,
                                                  const _Float16* __restrict__ Kb,
                                                  const _Float16* __restrict__ VT,
                                                  _Float16* __restrict__ O) {
  __shared__ __attribute__((aligned(16))) _Float16 Pl[4 * 16 * 72];

  const int wid = threadIdx.x >> 5;
  const int lane = threadIdx.x & 31;
  const int b = blockIdx.x / N_HEADS;
  const int h = blockIdx.x % N_HEADS;
  const int q0 = blockIdx.y * 64;
  const int col = lane & 15;
  const int rb = (lane >> 4) << 3;

  // Q fragments, pre-scaled by (1/sqrt(Hd)) * log2(e) once.
  const _Float16* Qbase =
      Q + (size_t)(b * SEQ + q0 + wid * 16) * D_MODEL + h * HEAD_DIM;
  v16h qa0 = frag_a(Qbase, D_MODEL);
  v16h qa1 = frag_a(Qbase + 32, D_MODEL);
  const _Float16 hsc = (_Float16)0.1803368801111244f;
#pragma unroll
  for (int i = 0; i < 16; ++i) { qa0[i] *= hsc; qa1[i] *= hsc; }

  v8f o[4] = {{}, {}, {}, {}};
  float m[8], l[8];
#pragma unroll
  for (int i = 0; i < 8; ++i) { m[i] = -3.0e38f; l[i] = 0.0f; }

  const _Float16* VTb = VT + (size_t)(b * N_HEADS + h) * HEAD_DIM * SEQ;
  const int nch = q0 / 64 + 1;
  _Float16* Pw = Pl + wid * 16 * 72;

  for (int c = 0; c < nch; ++c) {
    const int kc = c * 64;

    // Scores S = (Q*sc) * K^T; K rows contiguous in hd -> B-fragments.
    const _Float16* Kbase = Kb + (size_t)(b * SEQ + kc) * D_MODEL + h * HEAD_DIM;
    if (c + 1 < nch)  // prefetch next K chunk -> global_prefetch_b8
      __builtin_prefetch(Kbase + (size_t)(64 + (lane & 15)) * D_MODEL, 0, 0);
    v8f s[4];
#pragma unroll
    for (int t = 0; t < 4; ++t) {
      v8f z = {};
      z = wmma32(qa0, frag_b(Kbase + (size_t)t * 16 * D_MODEL, D_MODEL), z);
      z = wmma32(qa1, frag_b(Kbase + (size_t)t * 16 * D_MODEL + 32, D_MODEL), z);
      s[t] = z;
    }

    // Causal mask only on the diagonal chunk (kc == q0)
    if (kc == q0) {
      const int rowr = wid * 16 + rb;
#pragma unroll
      for (int t = 0; t < 4; ++t)
#pragma unroll
        for (int i = 0; i < 8; ++i)
          s[t][i] = ((t * 16 + col) > (rowr + i)) ? -3.0e38f : s[t][i];
    }

    // Online softmax: row == (vgpr idx, lane-half); permlane16 butterflies.
    float al[8];
#pragma unroll
    for (int i = 0; i < 8; ++i) {
      float cm = fmaxf(fmaxf(s[0][i], s[1][i]), fmaxf(s[2][i], s[3][i]));
      cm = redmax16(cm);
      const float mn = fmaxf(m[i], cm);
      al[i] = EXP2(m[i] - mn);
      m[i] = mn;
    }

    // P = exp2(s - m); stage into LDS (A-fragment re-layout) + row sums
    float rs[8];
#pragma unroll
    for (int i = 0; i < 8; ++i) rs[i] = 0.0f;
#pragma unroll
    for (int t = 0; t < 4; ++t)
#pragma unroll
      for (int i = 0; i < 8; ++i) {
        const float p = EXP2(s[t][i] - m[i]);
        rs[i] += p;
        Pw[(rb + i) * 72 + t * 16 + col] = (_Float16)p;
      }
#pragma unroll
    for (int i = 0; i < 8; ++i) l[i] = l[i] * al[i] + redsum16(rs[i]);

    // Rescale running output
#pragma unroll
    for (int t = 0; t < 4; ++t)
#pragma unroll
      for (int i = 0; i < 8; ++i) o[t][i] *= al[i];

    // Same-wave LDS RAW: DS ops are in-order; fence the compiler + drain.
    asm volatile("s_wait_dscnt 0" ::: "memory");

    // O += P * V : P A-fragments from LDS, V^T B-fragments from global.
    const v16h pa0 = frag_a(Pw, 72);       // keys kc..kc+31
    const v16h pa1 = frag_a(Pw + 32, 72);  // keys kc+32..kc+63
#pragma unroll
    for (int t = 0; t < 4; ++t) {
      o[t] = wmma32(pa0, frag_b(VTb + (size_t)t * 16 * SEQ + kc, SEQ), o[t]);
      o[t] = wmma32(pa1, frag_b(VTb + (size_t)t * 16 * SEQ + kc + 32, SEQ), o[t]);
    }
  }

  // Epilogue: O /= l, store f16 [tok][h*64 + hd]
  float inv[8];
#pragma unroll
  for (int i = 0; i < 8; ++i) inv[i] = 1.0f / l[i];
  _Float16* Ob = O + (size_t)(b * SEQ + q0 + wid * 16) * D_MODEL + h * HEAD_DIM;
#pragma unroll
  for (int t = 0; t < 4; ++t)
#pragma unroll
    for (int i = 0; i < 8; ++i)
      Ob[(size_t)(rb + i) * D_MODEL + t * 16 + col] = (_Float16)(o[t][i] * inv[i]);
}

// ---------------------------------------------------------------------------
extern "C" void kernel_launch(void* const* d_in, const int* in_sizes, int n_in,
                              void* d_out, int out_size, void* d_ws, size_t ws_size,
                              hipStream_t stream) {
  const float* x = (const float*)d_in[0];
  const float* Wq = (const float*)d_in[1];
  const float* Wk = (const float*)d_in[2];
  const float* Wv = (const float*)d_in[3];
  const float* Wo = (const float*)d_in[4];

  const size_t M1 = 1024u * 1024u;  // 1M halves
  _Float16* ws = (_Float16*)d_ws;
  _Float16* xh = ws;             // 4M halves
  _Float16* wqh = ws + 4 * M1;   // 1M each
  _Float16* wkh = ws + 5 * M1;
  _Float16* wvh = ws + 6 * M1;
  _Float16* woh = ws + 7 * M1;
  _Float16* Qb = ws + 8 * M1;    // 4M each
  _Float16* Kb = ws + 12 * M1;
  _Float16* Vt = ws + 16 * M1;   // V stored transposed per head: [B][H][64][S]
  _Float16* Ob = ws + 20 * M1;   // ends at 24M halves = 48 MB

  // Casts
  cast_f32_f16<<<(int)(4 * M1 / 4 / 256), 256, 0, stream>>>(x, xh, (int)(4 * M1));
  cast_f32_f16<<<(int)(M1 / 4 / 256), 256, 0, stream>>>(Wq, wqh, (int)M1);
  cast_f32_f16<<<(int)(M1 / 4 / 256), 256, 0, stream>>>(Wk, wkh, (int)M1);
  cast_f32_f16<<<(int)(M1 / 4 / 256), 256, 0, stream>>>(Wv, wvh, (int)M1);
  cast_f32_f16<<<(int)(M1 / 4 / 256), 256, 0, stream>>>(Wo, woh, (int)M1);

  // Projections: Q, K row-major; V stored transposed per head.
  dim3 gg(D_MODEL / 64, NTOK / 128);
  gemm_xwt<OUT_F16><<<gg, 256, 0, stream>>>(xh, wqh, (void*)Qb);
  gemm_xwt<OUT_F16><<<gg, 256, 0, stream>>>(xh, wkh, (void*)Kb);
  gemm_xwt<OUT_VT><<<gg, 256, 0, stream>>>(xh, wvh, (void*)Vt);

  // Attention
  dim3 ga(BATCH * N_HEADS, SEQ / 64);
  flash_attn<<<ga, 128, 0, stream>>>(Qb, Kb, Vt, Ob);

  // Output projection (f32 out)
  gemm_xwt<OUT_F32><<<gg, 256, 0, stream>>>(Ob, woh, d_out);
}
